// UnimodalBranch_2070174237004
// MI455X (gfx1250) — matrix operand order repaired and checked by reading, not compiled, not run on hardware
//
#include <hip/hip_runtime.h>
#include <cstddef>

typedef __attribute__((ext_vector_type(16))) _Float16 v16h;
typedef __attribute__((ext_vector_type(8)))  _Float16 v8h;
typedef __attribute__((ext_vector_type(8)))  float    v8f;

#define B_    8
#define CIN_  128
#define COUT_ 128
#define H_    64
#define W_    64
#define NPIX  (B_*H_*W_)     /* 32768 */
#define N_    50000
#define NV_   200000
#define M_    800000

// LDS A-tile: 3 rows x 18 cols x 128 cin (f16), row stride padded to 136
// halfs (272B = 68 dwords) so fragment reads spread across banks.
#define LDS_PIX_STRIDE 136
#define LDS_HALFS      (3 * 18 * LDS_PIX_STRIDE)

// ---------- workspace layout ----------
// [0]         x_nhwc  f16  NPIX*CIN_    = 8,388,608 B
// [8,388,608] w_pack  f16  9*COUT_*CIN_ =   294,912 B
// [8,683,520] feat    f32  NPIX*COUT_   = 16,777,216 B   (~25.5 MB total)
#define WS_XH_OFF   0
#define WS_WP_OFF   8388608
#define WS_FEAT_OFF 8683520

// ---------------------------------------------------------------
// Kernel 1: NCHW f32 -> NHWC f16 (writes coalesced over channel)
// ---------------------------------------------------------------
__global__ void nchw_to_nhwc_f16(const float* __restrict__ in,
                                 _Float16* __restrict__ out) {
    int idx = blockIdx.x * 256 + threadIdx.x;       // index into NHWC output
    int c = idx & 127;
    int p = idx >> 7;                               // pixel = ((b*H)+h)*W + w
    int w = p & 63;
    int h = (p >> 6) & 63;
    int b = p >> 12;
    out[idx] = (_Float16)in[((b * CIN_ + c) * H_ + h) * W_ + w];
}

// ---------------------------------------------------------------
// Kernel 2: conv_w [COUT][CIN][3][3] f32 -> w_pack [tap][cout][cin] f16
// ---------------------------------------------------------------
__global__ void repack_w_f16(const float* __restrict__ w,
                             _Float16* __restrict__ out) {
    int idx = blockIdx.x * 256 + threadIdx.x;       // (tap*128 + cout)*128 + cin
    if (idx >= 9 * COUT_ * CIN_) return;
    int cin  = idx & 127;
    int cout = (idx >> 7) & 127;
    int tap  = idx >> 14;
    out[idx] = (_Float16)w[(cout * CIN_ + cin) * 9 + tap];
}

// ---------------------------------------------------------------
// Kernel 3: implicit-GEMM 3x3 conv via v_wmma_f32_16x16x32_f16
//   block = 16 pixels (one M-tile, same image row) x 128 couts, 8 waves.
//   A halo tile staged once into LDS (zero-padded borders) -> the 36-WMMA
//   main loop is divergence-free; A comes from ds_load_b128.
// ---------------------------------------------------------------
__global__ void conv3x3_wmma(const _Float16* __restrict__ xh,   // NHWC f16
                             const _Float16* __restrict__ wp,   // [tap][cout][cin] f16
                             const float*    __restrict__ bias,
                             float*          __restrict__ feat) // [NPIX][COUT] f32
{
    __shared__ _Float16 a_lds[LDS_HALFS];

    const int tid  = threadIdx.x;
    const int lane = tid & 31;
    const int wv   = tid >> 5;                  // 0..7 -> cout tile
    const int m    = lane & 15;
    const int hi   = lane >> 4;

    const int p0 = blockIdx.x * 16;             // first pixel of M-tile
    const int b  = p0 >> 12;
    const int h  = (p0 >> 6) & 63;
    const int w0 = p0 & 63;

    // ---- cooperative halo stage: 3 rows x 18 cols x 16 v8h chunks = 864 ----
    for (int t = tid; t < 3 * 18 * 16; t += 256) {
        const int cc  = t & 15;                 // 8-half chunk within cin
        const int col = (t >> 4) % 18;          // 0..17  (w0-1 .. w0+16)
        const int r   = (t >> 4) / 18;          // 0..2   (h-1 .. h+1)
        const int hh  = h + r - 1;
        const int ww  = w0 + col - 1;
        v8h val = {};
        if (hh >= 0 && hh < H_ && ww >= 0 && ww < W_) {
            val = *(const v8h*)(xh + (size_t)((b * H_ + hh) * W_ + ww) * CIN_ + cc * 8);
        }
        *(v8h*)(a_lds + (r * 18 + col) * LDS_PIX_STRIDE + cc * 8) = val;
    }
    __syncthreads();

    const int coutb = wv * 16;
    const int cout  = coutb + m;                // lane's B column

    v8f acc = {};

    #pragma unroll
    for (int kh = 0; kh < 3; ++kh) {
        #pragma unroll
        for (int kw = 0; kw < 3; ++kw) {
            const _Float16* lrow = a_lds + (kh * 18 + m + kw) * LDS_PIX_STRIDE;
            const _Float16* wrow = wp + (size_t)((kh * 3 + kw) * COUT_ + cout) * CIN_;
            #pragma unroll
            for (int kc = 0; kc < 4; ++kc) {    // K = 128 = 4 chunks of 32
                // ISA 16-bit A layout: half-wave hi covers K [hi*8, hi*8+8)
                // and [16+hi*8, 16+hi*8+8) of each 32-wide K chunk.
                v8h lo  = *(const v8h*)(lrow + kc * 32 + hi * 8);
                v8h hi8 = *(const v8h*)(lrow + kc * 32 + 16 + hi * 8);
                v16h a;
                #pragma unroll
                for (int t = 0; t < 8; ++t) { a[t] = lo[t]; a[8 + t] = hi8[t]; }
                // B layout matches A: half-wave hi holds K [hi*8, hi*8+8)
                // and [16+hi*8, 16+hi*8+8) of column n.
                v8h blo = *(const v8h*)(wrow + kc * 32 + hi * 8);
                v8h bhi = *(const v8h*)(wrow + kc * 32 + 16 + hi * 8);
                v16h bf;
                #pragma unroll
                for (int t = 0; t < 8; ++t) { bf[t] = blo[t]; bf[8 + t] = bhi[t]; }
                acc = __builtin_amdgcn_wmma_f32_16x16x32_f16(
                        false, a, false, bf, (short)0, acc, false, false);
            }
        }
    }

    const float bv = bias[cout];                // C/D: n == lane&15 == m
    #pragma unroll
    for (int r = 0; r < 8; ++r) {
        const int mo = r + 8 * hi;              // C/D layout: VGPR r -> M = r + 8*hi
        feat[(size_t)(p0 + mo) * COUT_ + coutb + (lane & 15)] = acc[r] + bv;
    }
}

// ---------------------------------------------------------------
// Kernel 4: fused gather + atomic-CSR max + view-CSR max + add
//   one wave per 3D point; lane owns 4 channels (float4).
//   Index fetch software-pipelined ahead of the feat-row load.
// ---------------------------------------------------------------
__device__ __forceinline__ float4 f4max(float4 a, float4 b) {
    return make_float4(fmaxf(a.x, b.x), fmaxf(a.y, b.y),
                       fmaxf(a.z, b.z), fmaxf(a.w, b.w));
}

__global__ void gather_pool_fuse(const float* __restrict__ feat,
                                 const int*   __restrict__ pix_idx,
                                 const int*   __restrict__ atomic_ptr,
                                 const int*   __restrict__ view_ptr,
                                 const float* __restrict__ x3d,
                                 float*       __restrict__ out) {
    const int lane = threadIdx.x & 31;
    const int wv   = threadIdx.x >> 5;
    const int i    = blockIdx.x * 8 + wv;
    if (i >= N_) return;

    const int vs = view_ptr[i];
    const int ve = view_ptr[i + 1];
    const float NEG = -__builtin_inff();
    float4 outer = make_float4(NEG, NEG, NEG, NEG);

    for (int v = vs; v < ve; ++v) {
        const int js = atomic_ptr[v];
        const int je = atomic_ptr[v + 1];
        float4 inner = make_float4(NEG, NEG, NEG, NEG);
        int j = js;
        if (j < je) {
            int row = pix_idx[j];
            for (; j + 1 < je; ++j) {
                const int nrow = pix_idx[j + 1];          // prefetch next index
                const float4 f = *(const float4*)(feat + (size_t)row * COUT_ + lane * 4);
                inner = f4max(inner, f);
                row = nrow;
            }
            const float4 f = *(const float4*)(feat + (size_t)row * COUT_ + lane * 4);
            inner = f4max(inner, f);
        } else {
            inner = make_float4(0.f, 0.f, 0.f, 0.f);      // empty seg -> 0
        }
        outer = f4max(outer, inner);
    }
    if (ve <= vs) outer = make_float4(0.f, 0.f, 0.f, 0.f); // empty view -> 0

    const float4 xv = *(const float4*)(x3d + (size_t)i * COUT_ + lane * 4);
    float4 res = make_float4(xv.x + outer.x, xv.y + outer.y,
                             xv.z + outer.z, xv.w + outer.w);
    *(float4*)(out + (size_t)i * COUT_ + lane * 4) = res;
}

// ---------------------------------------------------------------
extern "C" void kernel_launch(void* const* d_in, const int* in_sizes, int n_in,
                              void* d_out, int out_size, void* d_ws, size_t ws_size,
                              hipStream_t stream) {
    const float* x3d   = (const float*)d_in[0];   // [N, COUT]
    const float* modx  = (const float*)d_in[1];   // [B, CIN, H, W]
    const float* convw = (const float*)d_in[2];   // [COUT, CIN, 3, 3]
    const float* convb = (const float*)d_in[3];   // [COUT]
    const int*   pix   = (const int*)d_in[4];     // [M]
    const int*   aptr  = (const int*)d_in[5];     // [NV+1]
    const int*   vptr  = (const int*)d_in[6];     // [N+1]
    float*       out   = (float*)d_out;           // [N, COUT]

    char* ws = (char*)d_ws;
    _Float16* xh   = (_Float16*)(ws + WS_XH_OFF);
    _Float16* wpk  = (_Float16*)(ws + WS_WP_OFF);
    float*    feat = (float*)(ws + WS_FEAT_OFF);

    // 1) NCHW -> NHWC f16
    nchw_to_nhwc_f16<<<(NPIX * CIN_) / 256, 256, 0, stream>>>(modx, xh);
    // 2) weight repack -> [tap][cout][cin] f16
    repack_w_f16<<<(9 * COUT_ * CIN_ + 255) / 256, 256, 0, stream>>>(convw, wpk);
    // 3) implicit-GEMM conv (WMMA, LDS-staged halo)
    conv3x3_wmma<<<NPIX / 16, 256, 0, stream>>>(xh, wpk, convb, feat);
    // 4) fused gather + double CSR max + add
    gather_pool_fuse<<<(N_ + 7) / 8, 256, 0, stream>>>(feat, pix, aptr, vptr, x3d, out);
}
